// GCGP_70660801954331
// MI455X (gfx1250) — compile-verified, matrix operand order checked by reading, not run on hardware
//
#include <hip/hip_runtime.h>
#include <hip/hip_bf16.h>

typedef __attribute__((ext_vector_type(16))) __bf16 v16bf;
typedef __attribute__((ext_vector_type(8)))  __bf16 v8bf;
typedef __attribute__((ext_vector_type(8)))  float  v8f;

// ---------------- graph prep ----------------

__global__ void count_kernel(const int* __restrict__ src, const int* __restrict__ dst,
                             int e, int* __restrict__ deg_in, int* __restrict__ deg_row) {
    int t = blockIdx.x * blockDim.x + threadIdx.x;
    if (t < e) {
        atomicAdd(&deg_in[dst[t]], 1);
        atomicAdd(&deg_row[src[t]], 1);
    }
}

__global__ void node_kernel(const int* __restrict__ deg_in, const int* __restrict__ deg_row,
                            float* __restrict__ dis, float* __restrict__ rowinv, int n) {
    int i = blockIdx.x * blockDim.x + threadIdx.x;
    if (i < n) {
        dis[i]    = rsqrtf((float)(deg_in[i] + 1));        // self-loop included
        rowinv[i] = 1.0f / ((float)deg_row[i] + 1.0f);     // final diagonal scale
    }
}

__global__ void scan_kernel(const int* __restrict__ cnt, int* __restrict__ off, int n) {
    if (threadIdx.x == 0 && blockIdx.x == 0) {
        int acc = 0;
        for (int i = 0; i < n; ++i) { off[i] = acc; acc += cnt[i]; }
        off[n] = acc;
    }
}

__global__ void fill_kernel(const int* __restrict__ src, const int* __restrict__ dst, int e,
                            const int* __restrict__ off, int* __restrict__ cursor,
                            const float* __restrict__ dis,
                            int* __restrict__ col, float* __restrict__ wgt) {
    int t = blockIdx.x * blockDim.x + threadIdx.x;
    if (t < e) {
        int s = src[t], d = dst[t];
        int p = atomicAdd(&cursor[d], 1);
        int j = off[d] + p;
        col[j] = s;
        wgt[j] = dis[s] * dis[d];
    }
}

// ---------------- bf16 hi/lo split + weighted squared norms ----------------

__global__ __launch_bounds__(256) void convert_kernel(
        const float* __restrict__ x, const float* __restrict__ w,
        __bf16* __restrict__ Ahi, __bf16* __restrict__ Alo,
        __bf16* __restrict__ Bhi, __bf16* __restrict__ Blo,
        float* __restrict__ sq, int d) {
    const int i = blockIdx.x;
    const int tid = threadIdx.x;
    float acc = 0.0f;
    for (int k = tid; k < d; k += 256) {
        size_t idx = (size_t)i * d + k;
        float xv = x[idx];
        float wv = w[k];
        float xwv = xv * wv;
        __bf16 bh = (__bf16)xv;  __bf16 bl = (__bf16)(xv  - (float)bh);
        __bf16 ah = (__bf16)xwv; __bf16 al = (__bf16)(xwv - (float)ah);
        Ahi[idx] = ah; Alo[idx] = al; Bhi[idx] = bh; Blo[idx] = bl;
        acc += xwv * xv;
    }
    __shared__ float red[256];
    red[tid] = acc;
    __syncthreads();
    for (int s = 128; s > 0; s >>= 1) {
        if (tid < s) red[tid] += red[tid + s];
        __syncthreads();
    }
    if (tid == 0) sq[i] = red[0];
}

// ---------------- WMMA GEMM + RBF epilogue ----------------
// A 16-bit 16x32 fragment (ISA 7.12.2): lane L (m = L%16, h = L/16) holds
// K = [8h, 8h+8) in elems 0..7 and K = [16+8h, 24+8h) in elems 8..15.
__device__ inline v16bf load_frag(const __bf16* __restrict__ p) {
    v8bf lo = *(const v8bf*)(p);
    v8bf hi = *(const v8bf*)(p + 16);
    v16bf f;
#pragma unroll
    for (int e = 0; e < 8; ++e) { f[e] = lo[e]; f[e + 8] = hi[e]; }
    return f;
}

__global__ __launch_bounds__(32) void gemm_rbf_kernel(
        const __bf16* __restrict__ Ahi, const __bf16* __restrict__ Alo,
        const __bf16* __restrict__ Bhi, const __bf16* __restrict__ Blo,
        const float* __restrict__ sq, float* __restrict__ K, int n, int d) {
    const int lane = threadIdx.x & 31;
    const int r    = lane & 15;
    const int h8   = (lane >> 4) << 3;
    const int tm   = blockIdx.x << 5;     // 32 rows per wave
    const int tn   = blockIdx.y << 5;     // 32 cols per wave
    v8f acc[2][2] = {};

    for (int k0 = 0; k0 < d; k0 += 32) {
        const int kk = k0 + h8;
        v16bf aH[2], aL[2], bH[2], bL[2];
#pragma unroll
        for (int t = 0; t < 2; ++t) {
            size_t ar = (size_t)(tm + t * 16 + r) * d + kk;
            size_t br = (size_t)(tn + t * 16 + r) * d + kk;
            aH[t] = load_frag(Ahi + ar);
            aL[t] = load_frag(Alo + ar);
            bH[t] = load_frag(Bhi + br);
            bL[t] = load_frag(Blo + br);
        }
#pragma unroll
        for (int mi = 0; mi < 2; ++mi) {
#pragma unroll
            for (int ni = 0; ni < 2; ++ni) {
                acc[mi][ni] = __builtin_amdgcn_wmma_f32_16x16x32_bf16(
                    false, aH[mi], false, bH[ni], (short)0, acc[mi][ni], false, false);
                acc[mi][ni] = __builtin_amdgcn_wmma_f32_16x16x32_bf16(
                    false, aH[mi], false, bL[ni], (short)0, acc[mi][ni], false, false);
                acc[mi][ni] = __builtin_amdgcn_wmma_f32_16x16x32_bf16(
                    false, aL[mi], false, bH[ni], (short)0, acc[mi][ni], false, false);
            }
        }
    }

    // K[i,j] = sq[i] + sq[j] - 2*G[i,j]; diag = 1
#pragma unroll
    for (int mi = 0; mi < 2; ++mi) {
#pragma unroll
        for (int ni = 0; ni < 2; ++ni) {
            const int cg = tn + ni * 16 + r;
            const float sqj = sq[cg];
#pragma unroll
            for (int g = 0; g < 8; ++g) {
                const int rg = tm + mi * 16 + h8 + g;   // C layout: m = VGPR + 8*(lane/16)
                float v = sq[rg] + sqj - 2.0f * acc[mi][ni][g];
                if (rg == cg) v = 1.0f;
                K[(size_t)rg * n + cg] = v;
            }
        }
    }
}

// ---------------- APPNP hop (CSR gather, alpha = 0.5) ----------------

__global__ __launch_bounds__(256) void hop_kernel(
        const float* __restrict__ x0, const float* __restrict__ hold,
        float* __restrict__ hnew,
        const int* __restrict__ off, const int* __restrict__ col,
        const float* __restrict__ wgt, const float* __restrict__ dis, int n) {
    const int i = blockIdx.x;
    const int c = (blockIdx.y << 10) + (threadIdx.x << 2);
    const size_t rowb = (size_t)i * n + c;
    const float dsi = dis[i];
    const float sn  = dsi * dsi;                 // self-loop norm
    const float4 xv = *(const float4*)(x0 + rowb);
    const float4 hv = *(const float4*)(hold + rowb);
    float4 acc;
    acc.x = sn * hv.x; acc.y = sn * hv.y; acc.z = sn * hv.z; acc.w = sn * hv.w;
    const int e0 = off[i], e1 = off[i + 1];
    for (int e = e0; e < e1; ++e) {
        const int s = col[e];
        const float wv = wgt[e];
        const float4 hs = *(const float4*)(hold + (size_t)s * n + c);
        acc.x = fmaf(wv, hs.x, acc.x);
        acc.y = fmaf(wv, hs.y, acc.y);
        acc.z = fmaf(wv, hs.z, acc.z);
        acc.w = fmaf(wv, hs.w, acc.w);
    }
    float4 o;
    o.x = 0.5f * acc.x + 0.5f * xv.x;
    o.y = 0.5f * acc.y + 0.5f * xv.y;
    o.z = 0.5f * acc.z + 0.5f * xv.z;
    o.w = 0.5f * acc.w + 0.5f * xv.w;
    *(float4*)(hnew + rowb) = o;
}

// ---------------- transposes ----------------

__global__ __launch_bounds__(256) void transpose_kernel(
        const float* __restrict__ in, float* __restrict__ out, int n) {
    __shared__ float tile[32][33];
    const int tx = threadIdx.x, ty = threadIdx.y;
    const int x  = blockIdx.x * 32 + tx;
    const int y0 = blockIdx.y * 32;
#pragma unroll
    for (int j = 0; j < 32; j += 8)
        tile[ty + j][tx] = in[(size_t)(y0 + ty + j) * n + x];
    __syncthreads();
    const int ox  = blockIdx.y * 32 + tx;
    const int oy0 = blockIdx.x * 32;
#pragma unroll
    for (int j = 0; j < 32; j += 8)
        out[(size_t)(oy0 + ty + j) * n + ox] = tile[tx][ty + j];
}

// out[i,j] = in[j,i]; diagonal additionally scaled by 1/(deg_row[i]+1)
__global__ __launch_bounds__(256) void final_kernel(
        const float* __restrict__ in, float* __restrict__ out,
        const float* __restrict__ rowinv, int n) {
    __shared__ float tile[32][33];
    const int tx = threadIdx.x, ty = threadIdx.y;
    const int x  = blockIdx.x * 32 + tx;
    const int y0 = blockIdx.y * 32;
#pragma unroll
    for (int j = 0; j < 32; j += 8)
        tile[ty + j][tx] = in[(size_t)(y0 + ty + j) * n + x];
    __syncthreads();
    const int ox  = blockIdx.y * 32 + tx;
    const int oy0 = blockIdx.x * 32;
#pragma unroll
    for (int j = 0; j < 32; j += 8) {
        const int oy = oy0 + ty + j;
        float v = tile[tx][ty + j];
        if (oy == ox) v *= rowinv[ox];
        out[(size_t)oy * n + ox] = v;
    }
}

// ---------------- host orchestration ----------------

extern "C" void kernel_launch(void* const* d_in, const int* in_sizes, int n_in,
                              void* d_out, int out_size, void* d_ws, size_t ws_size,
                              hipStream_t stream) {
    const float* x  = (const float*)d_in[0];   // [N, D]
    const float* w  = (const float*)d_in[1];   // [D]
    const int*   ei = (const int*)d_in[2];     // [2, E]
    const int D_ = in_sizes[1];
    const int N_ = in_sizes[0] / D_;
    const int E_ = in_sizes[2] / 2;
    const int* src = ei;
    const int* dst = ei + E_;
    float* out = (float*)d_out;

    char* ws = (char*)d_ws;
    size_t o = 0;
    auto take = [&](size_t bytes) -> char* {
        char* p = ws + o;
        o += (bytes + 255) & ~(size_t)255;
        return p;
    };
    float*  buf0    = (float*)take((size_t)N_ * N_ * 4);
    float*  buf1    = (float*)take((size_t)N_ * N_ * 4);
    __bf16* Ahi     = (__bf16*)take((size_t)N_ * D_ * 2);
    __bf16* Alo     = (__bf16*)take((size_t)N_ * D_ * 2);
    __bf16* Bhi     = (__bf16*)take((size_t)N_ * D_ * 2);
    __bf16* Blo     = (__bf16*)take((size_t)N_ * D_ * 2);
    float*  sq      = (float*)take((size_t)N_ * 4);
    float*  dis     = (float*)take((size_t)N_ * 4);
    float*  rowinv  = (float*)take((size_t)N_ * 4);
    int*    deg_in  = (int*)take((size_t)N_ * 4);
    int*    deg_row = (int*)take((size_t)N_ * 4);
    int*    offs    = (int*)take((size_t)(N_ + 1) * 4);
    int*    cursor  = (int*)take((size_t)N_ * 4);
    int*    col     = (int*)take((size_t)E_ * 4);
    float*  wgt     = (float*)take((size_t)E_ * 4);

    hipMemsetAsync(deg_in,  0, (size_t)N_ * 4, stream);
    hipMemsetAsync(deg_row, 0, (size_t)N_ * 4, stream);
    hipMemsetAsync(cursor,  0, (size_t)N_ * 4, stream);

    count_kernel<<<(E_ + 255) / 256, 256, 0, stream>>>(src, dst, E_, deg_in, deg_row);
    node_kernel<<<(N_ + 255) / 256, 256, 0, stream>>>(deg_in, deg_row, dis, rowinv, N_);
    scan_kernel<<<1, 32, 0, stream>>>(deg_in, offs, N_);
    fill_kernel<<<(E_ + 255) / 256, 256, 0, stream>>>(src, dst, E_, offs, cursor, dis, col, wgt);

    convert_kernel<<<N_, 256, 0, stream>>>(x, w, Ahi, Alo, Bhi, Blo, sq, D_);
    dim3 gg(N_ / 32, N_ / 32);
    gemm_rbf_kernel<<<gg, 32, 0, stream>>>(Ahi, Alo, Bhi, Blo, sq, buf0, N_, D_);

    dim3 hg(N_, N_ / 1024);
    // Phase 1: x0 = K (buf0); hop 10 lands in d_out.
    {
        const float* x0 = buf0;
        const float* hold = buf0;
        float* tgt[2] = { buf1, out };
        for (int t = 0; t < 10; ++t) {
            float* hnew = tgt[t & 1];
            hop_kernel<<<hg, 256, 0, stream>>>(x0, hold, hnew, offs, col, wgt, dis, N_);
            hold = hnew;
        }
    }
    dim3 tg(N_ / 32, N_ / 32);
    transpose_kernel<<<tg, dim3(32, 8), 0, stream>>>(out, buf0, N_);   // x0 for phase 2
    // Phase 2: hop 10 lands in buf1.
    {
        const float* x0 = buf0;
        const float* hold = buf0;
        float* tgt[2] = { out, buf1 };
        for (int t = 0; t < 10; ++t) {
            float* hnew = tgt[t & 1];
            hop_kernel<<<hg, 256, 0, stream>>>(x0, hold, hnew, offs, col, wgt, dis, N_);
            hold = hnew;
        }
    }
    final_kernel<<<tg, dim3(32, 8), 0, stream>>>(buf1, out, rowinv, N_);
}
